// MoELayer_65807488910119
// MI455X (gfx1250) — matrix operand order repaired and checked by reading, not compiled, not run on hardware
//
#include <hip/hip_runtime.h>

// ---------------------------------------------------------------- constants
static constexpr int B_ = 65536;
static constexpr int D_ = 256;
static constexpr int E_ = 16;
static constexpr int H_ = 256;
static constexpr int O_ = 256;
static constexpr int CHUNK_ = 16384;   // tokens per gate chunk (bounds workspace)

// ---------------------------------------------------------------- types
typedef __attribute__((ext_vector_type(16))) __bf16 v16bf;
typedef __attribute__((ext_vector_type(8)))  float  f32x8;

struct alignas(16) U128 { unsigned int d[4]; };
union Frag { U128 q[2]; unsigned int u[8]; v16bf v; };   // 32 bytes = 16 bf16

__device__ __forceinline__ unsigned short f32_to_bf16(float f) {
  union { float f; unsigned u; } c; c.f = f;
  unsigned u = c.u;
  return (unsigned short)((u + 0x7FFFu + ((u >> 16) & 1u)) >> 16);  // RNE
}

__device__ __forceinline__ f32x8 wmma_bf16(const v16bf& a, const v16bf& b, const f32x8& c) {
  return __builtin_amdgcn_wmma_f32_16x16x32_bf16(false, a, false, b, (short)0, c, false, false);
}

// ---------------------------------------------------------------- prep kernels
__global__ void cvt_f32_bf16(const float* __restrict__ src,
                             unsigned short* __restrict__ dst, long n) {
  long i = blockIdx.x * (long)blockDim.x + threadIdx.x;
  long stride = (long)gridDim.x * blockDim.x;
  for (; i < n; i += stride) dst[i] = f32_to_bf16(src[i]);
}

// Pack [nmat][K][N] row-major f32 into per-(ktile,ntile) WMMA B-fragment order:
// block (kt,nt) holds 32x16 bf16; lane L's 16 values are
//   B[kt*32 + (j>>3)*16 + (L>>4)*8 + (j&7)][nt*16 + (L&15)], j=0..15   (ISA 7.12.2 B layout)
__global__ void pack_b(const float* __restrict__ src, unsigned short* __restrict__ dst,
                       int K, int N, int nmat) {
  long total = (long)nmat * K * N;
  long per = (long)K * N;
  int numNT = N >> 4;
  long i = blockIdx.x * (long)blockDim.x + threadIdx.x;
  long stride = (long)gridDim.x * blockDim.x;
  for (; i < total; i += stride) {
    long mat = i / per;
    long r   = i % per;
    int  blk  = (int)(r >> 9);
    int  lane = (int)((r >> 4) & 31);
    int  j    = (int)(r & 15);
    int  kt = blk / numNT, nt = blk % numNT;
    int  k = kt * 32 + ((j >> 3) << 4) + ((lane >> 4) << 3) + (j & 7);
    int  n = nt * 16 + (lane & 15);
    dst[i] = f32_to_bf16(src[mat * per + (size_t)k * N + n]);
  }
}

// b1_eff[e][h] = b1[e][h] + sum_d bias[e][d] * W1[e][d][h]
__global__ void b1eff_kernel(const float* __restrict__ bias, const float* __restrict__ W1,
                             const float* __restrict__ b1, float* __restrict__ out) {
  int i = blockIdx.x * blockDim.x + threadIdx.x;
  if (i >= E_ * H_) return;
  int e = i / H_, h = i % H_;
  float acc = b1[i];
  const float* be = bias + (size_t)e * D_;
  const float* w  = W1 + (size_t)e * D_ * H_ + h;
  for (int d = 0; d < D_; ++d) acc += be[d] * w[(size_t)d * H_];
  out[i] = acc;
}

// ---------------------------------------------------------------- generic WMMA GEMM
// C[M,N] = act(A * Bpacked + bias). One wave computes an (MB*16) x (NT*16) tile:
// MB A-fragments are reused across NT B-fragments -> MB*NT wmma per MB+NT loads.
template <int MB, int NT>
__global__ void __launch_bounds__(256)
gemm_bf16(const unsigned short* __restrict__ A, const unsigned short* __restrict__ Bp,
          const float* __restrict__ bias, void* __restrict__ Out,
          int M, int N, int ldA, int ktiles, int relu, int outBf16, int ldOut) {
  const int lane = threadIdx.x & 31;
  const int wave = blockIdx.x * (blockDim.x >> 5) + (threadIdx.x >> 5);
  const int wavesPerRow = N / (16 * NT);
  const int mbase  = (wave / wavesPerRow) * (16 * MB);
  const int ntBase = (wave % wavesPerRow) * NT;
  if (mbase >= M) return;
  const int numNT = N >> 4;
  const int half = lane >> 4;
  const int col  = lane & 15;

  f32x8 acc[MB][NT] = {};
  const unsigned short* aRow[MB];
#pragma unroll
  for (int mb = 0; mb < MB; ++mb)
    aRow[mb] = A + (size_t)(mbase + mb * 16 + col) * ldA + half * 8;

  for (int kt = 0; kt < ktiles; ++kt) {
    Frag a[MB];
#pragma unroll
    for (int mb = 0; mb < MB; ++mb) {
      a[mb].q[0] = *(const U128*)(aRow[mb] + kt * 32);
      a[mb].q[1] = *(const U128*)(aRow[mb] + kt * 32 + 16);
    }
#pragma unroll
    for (int t = 0; t < NT; ++t) {
      const unsigned short* bp = Bp + ((size_t)(kt * numNT + ntBase + t) << 9) + lane * 16;
      Frag b;
      b.q[0] = *(const U128*)bp;
      b.q[1] = *(const U128*)(bp + 8);
#pragma unroll
      for (int mb = 0; mb < MB; ++mb)
        acc[mb][t] = wmma_bf16(a[mb].v, b.v, acc[mb][t]);
    }
  }

#pragma unroll
  for (int t = 0; t < NT; ++t) {
    int n = (ntBase + t) * 16 + col;
    float bv = bias[n];
#pragma unroll
    for (int mb = 0; mb < MB; ++mb) {
#pragma unroll
      for (int v = 0; v < 8; ++v) {
        int m = mbase + mb * 16 + v + half * 8;   // C/D layout: M = v + 8*half
        float val = acc[mb][t][v] + bv;
        if (relu) val = fmaxf(val, 0.f);
        if (outBf16) ((unsigned short*)Out)[(size_t)m * ldOut + n] = f32_to_bf16(val);
        else         ((float*)Out)[(size_t)m * ldOut + n] = val;
      }
    }
  }
}

// ---------------------------------------------------------------- softmax + top-2
__global__ void gate_softmax_topk(const float* __restrict__ logits,
                                  float* __restrict__ pOut, int nTok, long tokBase) {
  int t = blockIdx.x * blockDim.x + threadIdx.x;
  if (t >= nTok) return;
  float l[E_];
  float mx = -1e30f;
#pragma unroll
  for (int j = 0; j < E_; ++j) { l[j] = logits[(size_t)t * E_ + j]; mx = fmaxf(mx, l[j]); }
  float s = 0.f;
#pragma unroll
  for (int j = 0; j < E_; ++j) { l[j] = __expf(l[j] - mx); s += l[j]; }
  int i1 = 0; float v1 = l[0];
#pragma unroll
  for (int j = 1; j < E_; ++j) if (l[j] > v1) { v1 = l[j]; i1 = j; }
  int i2 = -1; float v2 = -1.f;
#pragma unroll
  for (int j = 0; j < E_; ++j) if (j != i1 && l[j] > v2) { v2 = l[j]; i2 = j; }
  float p1 = v1 / s, p2 = v2 / s;
  float inv = 1.f / (p1 + p2 + 1e-9f);
  float* dst = pOut + (size_t)(tokBase + t) * E_;
#pragma unroll
  for (int j = 0; j < E_; ++j)
    dst[j] = (j == i1) ? p1 * inv : ((j == i2) ? p2 * inv : 0.f);
}

// ---------------------------------------------------------------- fused expert kernel
// Block = 256 threads = 8 waves: (mw in 0..1)x(nw in 0..3); 64 tokens x 256 outputs.
// Wave tile = 32 tokens x 64 outputs (2 M-frags x 4 N-frags -> 8 wmma per 6 loads).
// Per expert: HE = p * relu(x@W1 + b1eff)  -> LDS bf16 (p folded into rows), then
// GEMM2 accumulates (p.he)@W2 DIRECTLY into persistent yacc; y += p*b2 scalar FMAs.
__global__ void __launch_bounds__(256)
moe_expert_kernel(const unsigned short* __restrict__ Xb,    // [B,D] bf16
                  const unsigned short* __restrict__ W1p,   // packed [E][8kt][16nt]
                  const unsigned short* __restrict__ W2p,   // packed [E][8kt][16nt]
                  const float* __restrict__ b1eff,          // [E,H]
                  const float* __restrict__ b2,             // [E,O]
                  const float* __restrict__ P,              // [B,E]
                  float* __restrict__ Y) {                  // [B,O]
  __shared__ unsigned short he[2][64 * 256];                // double-buffered, 64 KB
  const int lane = threadIdx.x & 31;
  const int w    = threadIdx.x >> 5;
  const int mw   = w >> 2;            // 32-token band (0..1)
  const int nw   = w & 3;             // 64-wide output band (0..3)
  const int half = lane >> 4;
  const int col  = lane & 15;
  const int tokBase = blockIdx.x * 64;

  f32x8 yacc[2][4] = {};              // persistent across experts
  const unsigned short* aRow[2];
#pragma unroll
  for (int mt = 0; mt < 2; ++mt)
    aRow[mt] = Xb + (size_t)(tokBase + mw * 32 + mt * 16 + col) * D_ + half * 8;

  for (int e = 0; e < E_; ++e) {
    // ---- phase 1: hacc = x @ W1_e
    f32x8 hacc[2][4] = {};
    for (int kt = 0; kt < 8; ++kt) {
      Frag a[2];
#pragma unroll
      for (int mt = 0; mt < 2; ++mt) {
        a[mt].q[0] = *(const U128*)(aRow[mt] + kt * 32);
        a[mt].q[1] = *(const U128*)(aRow[mt] + kt * 32 + 16);
      }
#pragma unroll
      for (int t = 0; t < 4; ++t) {
        const unsigned short* bp =
            W1p + ((size_t)((e * 8 + kt) * 16 + nw * 4 + t) << 9) + lane * 16;
        Frag b;
        b.q[0] = *(const U128*)bp;
        b.q[1] = *(const U128*)(bp + 8);
#pragma unroll
        for (int mt = 0; mt < 2; ++mt)
          hacc[mt][t] = wmma_bf16(a[mt].v, b.v, hacc[mt][t]);
      }
    }
    // gate weights for this wave's 32 token rows
    float pv[2][8];
#pragma unroll
    for (int mt = 0; mt < 2; ++mt)
#pragma unroll
      for (int v = 0; v < 8; ++v) {
        int tok = tokBase + mw * 32 + mt * 16 + v + half * 8;
        pv[mt][v] = P[(size_t)tok * E_ + e];
      }
    // relu + b1eff, scale rows by p, write bf16 to LDS; fold p*b2 into yacc now
#pragma unroll
    for (int t = 0; t < 4; ++t) {
      int hc = nw * 64 + t * 16 + col;
      float bv  = b1eff[e * H_ + hc];
      float b2v = b2[e * O_ + hc];
#pragma unroll
      for (int mt = 0; mt < 2; ++mt) {
#pragma unroll
        for (int v = 0; v < 8; ++v) {
          int m = mw * 32 + mt * 16 + v + half * 8;
          float hv = pv[mt][v] * fmaxf(hacc[mt][t][v] + bv, 0.f);
          he[e & 1][m * 256 + hc] = f32_to_bf16(hv);
          yacc[mt][t][v] += pv[mt][v] * b2v;
        }
      }
    }
    __syncthreads();   // double-buffered -> one barrier per expert suffices

    // ---- phase 2: yacc += (p.he) @ W2_e   (direct accumulation across experts)
    for (int kt = 0; kt < 8; ++kt) {
      Frag a[2];
#pragma unroll
      for (int mt = 0; mt < 2; ++mt) {
        int lbase = (mw * 32 + mt * 16 + col) * 256 + kt * 32 + half * 8;
        a[mt].q[0] = *(const U128*)&he[e & 1][lbase];
        a[mt].q[1] = *(const U128*)&he[e & 1][lbase + 16];
      }
#pragma unroll
      for (int t = 0; t < 4; ++t) {
        const unsigned short* bp =
            W2p + ((size_t)((e * 8 + kt) * 16 + nw * 4 + t) << 9) + lane * 16;
        Frag b;
        b.q[0] = *(const U128*)bp;
        b.q[1] = *(const U128*)(bp + 8);
#pragma unroll
        for (int mt = 0; mt < 2; ++mt)
          yacc[mt][t] = wmma_bf16(a[mt].v, b.v, yacc[mt][t]);
      }
    }
  }

#pragma unroll
  for (int t = 0; t < 4; ++t) {
    int n = nw * 64 + t * 16 + col;
#pragma unroll
    for (int mt = 0; mt < 2; ++mt) {
#pragma unroll
      for (int v = 0; v < 8; ++v) {
        int tok = tokBase + mw * 32 + mt * 16 + v + half * 8;
        Y[(size_t)tok * O_ + n] = yacc[mt][t][v];
      }
    }
  }
}

// ---------------------------------------------------------------- launcher
extern "C" void kernel_launch(void* const* d_in, const int* in_sizes, int n_in,
                              void* d_out, int out_size, void* d_ws, size_t ws_size,
                              hipStream_t stream) {
  (void)in_sizes; (void)n_in; (void)out_size; (void)ws_size;
  const float* x   = (const float*)d_in[0];
  const float* bias= (const float*)d_in[1];
  const float* W1  = (const float*)d_in[2];
  const float* b1  = (const float*)d_in[3];
  const float* W2  = (const float*)d_in[4];
  const float* b2  = (const float*)d_in[5];
  const float* g1  = (const float*)d_in[6];
  const float* gb1 = (const float*)d_in[7];
  const float* g2  = (const float*)d_in[8];
  const float* gb2 = (const float*)d_in[9];
  const float* g3  = (const float*)d_in[10];
  const float* gb3 = (const float*)d_in[11];

  float* y = (float*)d_out;                         // [B,O]
  float* p = (float*)d_out + (size_t)B_ * O_;       // [B,E]

  char* ws = (char*)d_ws;
  size_t off = 0;
  auto alloc = [&](size_t bytes) { void* r = ws + off; off += (bytes + 255) & ~(size_t)255; return r; };
  unsigned short* xb  = (unsigned short*)alloc((size_t)B_ * D_ * 2);        // 32 MB
  unsigned short* w1p = (unsigned short*)alloc((size_t)E_ * D_ * H_ * 2);   // 2 MB
  unsigned short* w2p = (unsigned short*)alloc((size_t)E_ * H_ * O_ * 2);   // 2 MB
  unsigned short* g1p = (unsigned short*)alloc((size_t)D_ * 4 * D_ * 2);    // 512 KB
  unsigned short* g2p = (unsigned short*)alloc((size_t)4 * D_ * D_ * 2);    // 512 KB
  unsigned short* g3p = (unsigned short*)alloc((size_t)D_ * E_ * 2);        // 8 KB
  float*          b1e = (float*)alloc((size_t)E_ * H_ * 4);                 // 16 KB
  unsigned short* h1  = (unsigned short*)alloc((size_t)CHUNK_ * 1024 * 2);  // 32 MB
  unsigned short* h2  = (unsigned short*)alloc((size_t)CHUNK_ * 256 * 2);   // 8 MB
  float*          lg  = (float*)alloc((size_t)CHUNK_ * E_ * 4);             // 1 MB

  // ---- prep: convert + pack weights into WMMA fragment order
  cvt_f32_bf16<<<4096, 256, 0, stream>>>(x, xb, (long)B_ * D_);
  pack_b<<<2048, 256, 0, stream>>>(W1, w1p, D_, H_, E_);
  pack_b<<<2048, 256, 0, stream>>>(W2, w2p, H_, O_, E_);
  pack_b<<<512, 256, 0, stream>>>(g1, g1p, D_, 4 * D_, 1);
  pack_b<<<512, 256, 0, stream>>>(g2, g2p, 4 * D_, D_, 1);
  pack_b<<<16, 256, 0, stream>>>(g3, g3p, D_, E_, 1);
  b1eff_kernel<<<(E_ * H_ + 255) / 256, 256, 0, stream>>>(bias, W1, b1, b1e);

  // ---- gate MLP + softmax/top2, chunked over tokens
  for (int c = 0; c < B_ / CHUNK_; ++c) {
    long tb = (long)c * CHUNK_;
    const unsigned short* xc = xb + (size_t)tb * D_;
    {   // h1 = relu(x @ g1 + gb1)           [CHUNK, 1024]
      int waves = (CHUNK_ / 32) * (1024 / 64);
      gemm_bf16<2, 4><<<waves / 8, 256, 0, stream>>>(xc, g1p, gb1, h1,
                                                     CHUNK_, 1024, D_, 8, 1, 1, 1024);
    }
    {   // h2 = relu(h1 @ g2 + gb2)          [CHUNK, 256]
      int waves = (CHUNK_ / 32) * (256 / 64);
      gemm_bf16<2, 4><<<waves / 8, 256, 0, stream>>>(h1, g2p, gb2, h2,
                                                     CHUNK_, 256, 1024, 32, 1, 1, 256);
    }
    {   // logits = h2 @ g3 + gb3            [CHUNK, 16] fp32
      int waves = (CHUNK_ / 32) * (16 / 16);
      gemm_bf16<2, 1><<<waves / 8, 256, 0, stream>>>(h2, g3p, gb3, lg,
                                                     CHUNK_, 16, 256, 8, 0, 0, 16);
    }
    gate_softmax_topk<<<CHUNK_ / 256, 256, 0, stream>>>(lg, p, CHUNK_, tb);
  }

  // ---- fused dense-expert compute + weighted combine
  moe_expert_kernel<<<B_ / 64, 256, 0, stream>>>(xb, w1p, w2p, b1e, b2, p, y);
}
